// MultiHeadAttention_66958540145106
// MI455X (gfx1250) — matrix operand order repaired and checked by reading, not compiled, not run on hardware
//
#include <hip/hip_runtime.h>

typedef __attribute__((ext_vector_type(16))) _Float16 v16h;
typedef __attribute__((ext_vector_type(8)))  _Float16 v8h;
typedef __attribute__((ext_vector_type(8)))  float    v8f;
typedef __attribute__((ext_vector_type(4)))  float    v4f;

typedef unsigned int u32;
typedef __attribute__((ext_vector_type(4))) u32 u32x4;
typedef __attribute__((ext_vector_type(8))) int i32x8;
typedef __attribute__((ext_vector_type(4))) int i32x4;

#define NB 4
#define NS 2048
#define NE 1024
#define NH 8
#define ND 128

static __device__ __forceinline__ v8f wmma16(v16h a, v16h b, v8f c) {
  // D = A(16x32 f16) * B(32x16 f16) + C(16x16 f32)
  return __builtin_amdgcn_wmma_f32_16x16x32_f16(false, a, false, b, (short)0, c,
                                                false, false);
}

// ---------------------------------------------------------------------------
// TDM: async 2-D tile load Global -> LDS via Tensor Data Mover.
// Descriptor per CDNA5 ISA (D# groups 0/1; groups 2/3
// zero => dims 2..4 unused). pad_interval: pad after 2^(v+1) DWORDs;
// pad_amount: (v+1) DWORDs of LDS padding (load only).
// ---------------------------------------------------------------------------
static __device__ __forceinline__ void tdm_load_2d(
    u32 lds_addr, unsigned long long gaddr,
    u32 tensor_d0, u32 tensor_d1, u32 tile_d0, u32 tile_d1,
    u32 d0_stride, u32 pad_interval, u32 pad_amount)
{
  u32x4 g0;
  g0[0] = 1u;                                        // count=1, user mode
  g0[1] = lds_addr;                                  // LDS byte address
  g0[2] = (u32)(gaddr & 0xFFFFFFFFu);                // global addr [31:0]
  g0[3] = (u32)((gaddr >> 32) & 0x1FFFFFFu)          // global addr [56:32]
        | (2u << 30);                                // type = 2 ("image")
  i32x8 g1;
  g1[0] = (int)((1u << 16)                           // data_size = 2 bytes
              | (1u << 20)                           // pad_enable
              | (pad_interval << 22)
              | (pad_amount << 25));
  g1[1] = (int)((tensor_d0 & 0xFFFFu) << 16);        // tensor_dim0 lo16
  g1[2] = (int)((tensor_d0 >> 16) | ((tensor_d1 & 0xFFFFu) << 16));
  g1[3] = (int)((tensor_d1 >> 16) | (tile_d0 << 16));// tile_dim0
  g1[4] = (int)tile_d1;                              // tile_dim1 (tile_dim2=0)
  g1[5] = (int)d0_stride;                            // tensor_dim0_stride lo32
  g1[6] = 0;
  g1[7] = 0;
  i32x4 gz = {0, 0, 0, 0};
#if __clang_major__ >= 23
  i32x8 gz8 = {0, 0, 0, 0, 0, 0, 0, 0};
  __builtin_amdgcn_tensor_load_to_lds(g0, g1, gz, gz, gz8, 0);
#else
  __builtin_amdgcn_tensor_load_to_lds(g0, g1, gz, gz, 0);
#endif
}

// ---------------------------------------------------------------------------
// Kernel 1: per-head QKV projection, f32 -> f16, WMMA GEMM.
//   Q gets the reference 1/sqrt(H) score scale folded in.
//   Q,K stored [B,H,S,D] f16 ; V stored transposed [B,H,D,S] f16.
// waves = 3 (QKV) * 8 (H) * 512 (row blocks of B*S=8192) = 12288
// ---------------------------------------------------------------------------
__global__ __launch_bounds__(128)
void qkv_proj_kernel(const float* __restrict__ x,
                     const float* __restrict__ Wq, const float* __restrict__ Wk,
                     const float* __restrict__ Wv,
                     const float* __restrict__ bq, const float* __restrict__ bk,
                     const float* __restrict__ bv,
                     _Float16* __restrict__ Qh, _Float16* __restrict__ Kh,
                     _Float16* __restrict__ Vt)
{
  const int lane = threadIdx.x & 31;
  const int wid  = threadIdx.x >> 5;
  const int gw   = blockIdx.x * 4 + wid;          // 0..12287
  const int rb   = gw & 511;
  const int hh   = (gw >> 9) & 7;
  const int m3   = gw >> 12;                      // 0=Q 1=K 2=V

  const float* W    = (m3 == 0) ? Wq : (m3 == 1) ? Wk : Wv;
  const float* bias = (m3 == 0) ? bq : (m3 == 1) ? bk : bv;

  const int m    = lane & 15;
  const int half = lane >> 4;

  const int   row  = rb * 16 + m;                 // flat (b,s) row
  const float* xrow = x + (size_t)row * NE + hh * ND;
  v16h a[4];
#pragma unroll
  for (int kc = 0; kc < 4; ++kc) {
    v4f lo0 = *(const v4f*)(xrow + kc * 32 + half * 16);
    v4f lo1 = *(const v4f*)(xrow + kc * 32 + half * 16 + 4);
    v4f hi0 = *(const v4f*)(xrow + kc * 32 + half * 16 + 8);
    v4f hi1 = *(const v4f*)(xrow + kc * 32 + half * 16 + 12);
#pragma unroll
    for (int e = 0; e < 4; ++e) {
      a[kc][e]      = (_Float16)lo0[e];
      a[kc][e + 4]  = (_Float16)lo1[e];
      a[kc][e + 8]  = (_Float16)hi0[e];
      a[kc][e + 12] = (_Float16)hi1[e];
    }
  }

  const float qscale = 0.35355339059327373f;      // 1/sqrt(H=8)
  const int   n = m;

#pragma unroll 1
  for (int oc = 0; oc < 8; ++oc) {
    float bb = bias[hh * ND + oc * 16 + n];
    v8f c;
#pragma unroll
    for (int r = 0; r < 8; ++r) c[r] = bb;

#pragma unroll
    for (int kc = 0; kc < 4; ++kc) {
      const float* wp = W + (size_t)hh * ND * ND + (size_t)(oc * 16 + n) * ND
                          + kc * 32 + half * 16;
      v16h bf;
#pragma unroll
      for (int q4 = 0; q4 < 4; ++q4) {
        v4f w4 = *(const v4f*)(wp + q4 * 4);
#pragma unroll
        for (int e = 0; e < 4; ++e) bf[q4 * 4 + e] = (_Float16)w4[e];
      }
      c = wmma16(a[kc], bf, c);
    }

    if (m3 == 2) {                                // V transposed: [B,H,D,S]
#pragma unroll
      for (int r = 0; r < 8; ++r) {
        int rowi = rb * 16 + r + 8 * half;
        int b2 = rowi >> 11, s2 = rowi & 2047;
        Vt[((size_t)(b2 * NH + hh) * ND + oc * 16 + n) * NS + s2] = (_Float16)c[r];
      }
    } else {                                      // Q/K: [B,H,S,D]
      _Float16* dst = (m3 == 0) ? Qh : Kh;
      float sc = (m3 == 0) ? qscale : 1.0f;
#pragma unroll
      for (int r = 0; r < 8; ++r) {
        int rowi = rb * 16 + r + 8 * half;
        int b2 = rowi >> 11, s2 = rowi & 2047;
        dst[((size_t)(b2 * NH + hh) * NS + s2) * ND + oc * 16 + n] =
            (_Float16)(c[r] * sc);
      }
    }
  }
}

// ---------------------------------------------------------------------------
// Kernel 2: flash attention. 4 waves/block share one (b,h); each wave owns a
// 16-row Q tile. K/V tiles (32 keys) are TDM-loaded into double-buffered LDS
// once per block per iteration; fragments read via ds_load_b128.
// K LDS row stride 272B (256B data + 16B TDM pad); V row stride 80B (64+16).
// ---------------------------------------------------------------------------
#define PROW 40            // P-tile LDS row stride (halves)
#define KROW 136           // K-tile LDS row stride (halves) = 272B
#define VROW 40            // V-tile LDS row stride (halves) = 80B

__global__ __launch_bounds__(128)
void flash_attn_kernel(const _Float16* __restrict__ Qh,
                       const _Float16* __restrict__ Kh,
                       const _Float16* __restrict__ Vt,
                       float* __restrict__ out)
{
  __shared__ _Float16 kbuf[2][32 * KROW];     // 2 x 8704 B
  __shared__ _Float16 vbuf[2][128 * VROW];    // 2 x 10240 B
  __shared__ _Float16 pbuf[4][16 * PROW];     // 4 x 1280 B

  const int lane = threadIdx.x & 31;
  const int wid  = threadIdx.x >> 5;
  const int bid  = blockIdx.x;                // 0..1023
  const int qg   = bid & 31;                  // group of 4 Q tiles
  const int hh   = (bid >> 5) & 7;
  const int b    = bid >> 8;
  const int qb   = qg * 4 + wid;
  const int m    = lane & 15;
  const int half = lane >> 4;

  const size_t bh = (size_t)(b * NH + hh);
  const _Float16* Qp = Qh + (bh * NS + (size_t)qb * 16) * ND;
  const unsigned long long kg = (unsigned long long)(uintptr_t)(Kh + bh * NS * ND);
  const unsigned long long vg = (unsigned long long)(uintptr_t)(Vt + bh * (size_t)ND * NS);
  const u32 kl0 = (u32)(uintptr_t)&kbuf[0][0];
  const u32 kl1 = (u32)(uintptr_t)&kbuf[1][0];
  const u32 vl0 = (u32)(uintptr_t)&vbuf[0][0];
  const u32 vl1 = (u32)(uintptr_t)&vbuf[1][0];
  _Float16* lp = pbuf[wid];

  // ---- Q A-fragments (registers, 4 x v16h) ----
  v16h aq[4];
#pragma unroll
  for (int kc = 0; kc < 4; ++kc) {
    v8h lo = *(const v8h*)(Qp + (size_t)m * ND + kc * 32 + half * 16);
    v8h hi = *(const v8h*)(Qp + (size_t)m * ND + kc * 32 + half * 16 + 8);
#pragma unroll
    for (int e = 0; e < 8; ++e) { aq[kc][e] = lo[e]; aq[kc][8 + e] = hi[e]; }
  }

  float mrun[8], lrun[8];
  v8f o[8];
#pragma unroll
  for (int r = 0; r < 8; ++r) { mrun[r] = -3.0e38f; lrun[r] = 0.0f; }
#pragma unroll
  for (int oc = 0; oc < 8; ++oc)
#pragma unroll
    for (int r = 0; r < 8; ++r) o[oc][r] = 0.0f;

  // ---- prologue: TDM-load tiles for j=0 into buffer 0 (wave 0 only) ----
  if (wid == 0) {
    // K tile: 32 rows x 128 halves, row 256B, pad 16B after each 64 DWORDs
    tdm_load_2d(kl0, kg, ND, NS, ND, 32, ND, 5u, 3u);
    // V tile: 128 rows x 32 halves, row 64B, pad 16B after each 16 DWORDs
    tdm_load_2d(vl0, vg, NS, ND, 32, ND, NS, 3u, 3u);
  }

#pragma unroll 1
  for (int j = 0; j < NS / 32; ++j) {
    if (wid == 0) {
      if (j < NS / 32 - 1) {
        const int nb = (j + 1) & 1;
        tdm_load_2d(nb ? kl1 : kl0, kg + (unsigned long long)(j + 1) * 32u * 256u,
                    ND, NS, ND, 32, ND, 5u, 3u);
        tdm_load_2d(nb ? vl1 : vl0, vg + (unsigned long long)(j + 1) * 64u,
                    NS, ND, 32, ND, NS, 3u, 3u);
        __builtin_amdgcn_s_wait_tensorcnt((short)2);  // current buffer done
      } else {
        __builtin_amdgcn_s_wait_tensorcnt((short)0);
      }
    }
    __syncthreads();                                  // tiles visible to all
    const _Float16* kt = kbuf[j & 1];
    const _Float16* vt = vbuf[j & 1];

    // ---- S = Q K^T : two 16x16 tiles (keys j*32 .. j*32+31) ----
    v8f s0, s1;
#pragma unroll
    for (int r = 0; r < 8; ++r) { s0[r] = 0.0f; s1[r] = 0.0f; }
#pragma unroll
    for (int kc = 0; kc < 4; ++kc) {
      v16h bk0 = *(const v16h*)(kt + (size_t)m * KROW + kc * 32 + half * 16);
      v16h bk1 = *(const v16h*)(kt + (size_t)(16 + m) * KROW + kc * 32 + half * 16);
      s0 = wmma16(aq[kc], bk0, s0);
      s1 = wmma16(aq[kc], bk1, s1);
    }

    // ---- online softmax (rows r+8*half live in this half's 16 lanes) ----
    float alpha[8];
#pragma unroll
    for (int r = 0; r < 8; ++r) {
      float mx = fmaxf(s0[r], s1[r]);
      mx = fmaxf(mx, __shfl_xor(mx, 1, 32));
      mx = fmaxf(mx, __shfl_xor(mx, 2, 32));
      mx = fmaxf(mx, __shfl_xor(mx, 4, 32));
      mx = fmaxf(mx, __shfl_xor(mx, 8, 32));
      float mnew = fmaxf(mrun[r], mx);
      float al   = __expf(mrun[r] - mnew);
      float p0   = __expf(s0[r] - mnew);
      float p1   = __expf(s1[r] - mnew);
      float rs   = p0 + p1;
      rs += __shfl_xor(rs, 1, 32);
      rs += __shfl_xor(rs, 2, 32);
      rs += __shfl_xor(rs, 4, 32);
      rs += __shfl_xor(rs, 8, 32);
      lrun[r] = lrun[r] * al + rs;
      mrun[r] = mnew;
      alpha[r] = al;
      s0[r] = p0; s1[r] = p1;
    }
#pragma unroll
    for (int oc = 0; oc < 8; ++oc)
#pragma unroll
      for (int r = 0; r < 8; ++r) o[oc][r] *= alpha[r];

    // ---- P: C-layout -> A-layout via per-wave LDS tile ----
#pragma unroll
    for (int r = 0; r < 8; ++r) {
      int rr = r + 8 * half;
      lp[rr * PROW + m]      = (_Float16)s0[r];
      lp[rr * PROW + 16 + m] = (_Float16)s1[r];
    }
    __asm__ volatile("s_wait_dscnt 0x0" ::: "memory");
    v16h ap;
    {
      v8h lo = *(const v8h*)(lp + m * PROW + half * 16);
      v8h hi = *(const v8h*)(lp + m * PROW + half * 16 + 8);
#pragma unroll
      for (int e = 0; e < 8; ++e) { ap[e] = lo[e]; ap[8 + e] = hi[e]; }
    }
    __asm__ volatile("" ::: "memory");

    // ---- O += P * V ----
#pragma unroll
    for (int oc = 0; oc < 8; ++oc) {
      v16h bv = *(const v16h*)(vt + (size_t)(oc * 16 + m) * VROW + half * 16);
      o[oc] = wmma16(ap, bv, o[oc]);
    }
    __syncthreads();                  // all waves done with this buffer
  }

  // ---- normalize and store ----
#pragma unroll
  for (int r = 0; r < 8; ++r) lrun[r] = 1.0f / lrun[r];
  float* orow = out + ((size_t)b * NS + (size_t)qb * 16) * NE + hh * ND;
#pragma unroll
  for (int oc = 0; oc < 8; ++oc)
#pragma unroll
    for (int r = 0; r < 8; ++r) {
      int rr = r + 8 * half;
      orow[(size_t)rr * NE + oc * 16 + m] = o[oc][r] * lrun[r];
    }
}

// ---------------------------------------------------------------------------
extern "C" void kernel_launch(void* const* d_in, const int* in_sizes, int n_in,
                              void* d_out, int out_size, void* d_ws, size_t ws_size,
                              hipStream_t stream) {
  const float* x  = (const float*)d_in[0];
  const float* Wq = (const float*)d_in[1];
  const float* Wk = (const float*)d_in[2];
  const float* Wv = (const float*)d_in[3];
  const float* bq = (const float*)d_in[4];
  const float* bk = (const float*)d_in[5];
  const float* bv = (const float*)d_in[6];
  float* out = (float*)d_out;

  const size_t mat = (size_t)NB * NH * NS * ND;   // 8M halves each
  _Float16* Qh = (_Float16*)d_ws;
  _Float16* Kh = Qh + mat;
  _Float16* Vt = Kh + mat;

  qkv_proj_kernel<<<3072, 128, 0, stream>>>(x, Wq, Wk, Wv, bq, bk, bv, Qh, Kh, Vt);
  flash_attn_kernel<<<1024, 128, 0, stream>>>(Qh, Kh, Vt, out);
}
